// EnhancedInformer_1194000908687
// MI455X (gfx1250) — matrix operand order repaired and checked by reading, not compiled, not run on hardware
//
#include <hip/hip_runtime.h>
#include <math.h>
#include <stdint.h>

typedef __bf16 bf16_t;
typedef __attribute__((ext_vector_type(16))) __bf16 v16bf;
typedef __attribute__((ext_vector_type(8)))  __bf16 v8bf;
typedef __attribute__((ext_vector_type(8)))  float  v8f;
typedef __attribute__((ext_vector_type(4)))  unsigned int v4u;
typedef __attribute__((ext_vector_type(8)))  int v8i;
typedef __attribute__((ext_vector_type(4)))  int v4i;

// ---------------------------------------------------------------------------
// WMMA helpers
// ---------------------------------------------------------------------------
__device__ __forceinline__ v8f wmma_bf16(v16bf a, v16bf b, v8f c) {
    return __builtin_amdgcn_wmma_f32_16x16x32_bf16(
        /*neg_a=*/false, a, /*neg_b=*/false, b,
        /*c_mod=*/(short)0, c, /*reuse_a=*/false, /*reuse_b=*/false);
}

// A fragment (16x32 bf16): lane holds row M=lane%16; elems = 16 contiguous K at half*16.
__device__ __forceinline__ v16bf load_frag_a(const bf16_t* base, int half) {
    v8bf lo = *reinterpret_cast<const v8bf*>(base + half * 16);
    v8bf hi = *reinterpret_cast<const v8bf*>(base + half * 16 + 8);
    return __builtin_shufflevector(lo, hi, 0,1,2,3,4,5,6,7,8,9,10,11,12,13,14,15);
}

// B fragment (32x16 bf16): lane holds col N=lane%16; elems = 16 contiguous K at half*16.
__device__ __forceinline__ v16bf load_frag_b(const bf16_t* base, int half) {
    v8bf lo = *reinterpret_cast<const v8bf*>(base + half * 16);
    v8bf hi = *reinterpret_cast<const v8bf*>(base + half * 16 + 8);
    return __builtin_shufflevector(lo, hi, 0,1,2,3,4,5,6,7,8,9,10,11,12,13,14,15);
}

__device__ __forceinline__ float gelu_exact(float x) {
    return 0.5f * x * (1.0f + erff(x * 0.70710678118654752f));
}

// ---------------------------------------------------------------------------
// TDM: DMA one [128 rows x 32 cols] bf16 tile (row stride K elems) from global
// into LDS, writing LDS rows padded to 48 bf16 (16 DWORDs data + 8 DWORDs pad).
// Issued once per block by wave 0; tracked by TENSORcnt.
// ---------------------------------------------------------------------------
__device__ __forceinline__ void tdm_load_w_tile(const bf16_t* gptr,
                                                unsigned lds_off, int K) {
    unsigned long long ga = (unsigned long long)(uintptr_t)gptr;
    v4u g0;
    g0[0] = 1u;                                     // count=1, user descriptor
    g0[1] = lds_off;                                // LDS byte address
    g0[2] = (unsigned)(ga & 0xFFFFFFFFull);         // global_addr[31:0]
    g0[3] = (unsigned)((ga >> 32) & 0x01FFFFFFull)  // global_addr[56:32]
          | (2u << 30);                             // type = 2 ("image")
    v8i g1;
    g1[0] = (1 << 16)      // data_size = 1 -> 2 bytes
          | (1 << 20)      // pad_enable
          | (3 << 22)      // pad_interval: 16 DWORDs (64B) of data ...
          | (7 << 25);     // pad_amount:   ... then 8 DWORDs (32B) pad
    g1[1] = (K & 0xFFFF) << 16;   // tensor_dim0[15:0] = K   (atomic_barrier_addr=0)
    g1[2] = (128 << 16);          // tensor_dim0[31:16]=0 | tensor_dim1 = 128
    g1[3] = (32 << 16);           // tensor_dim1 hi =0 | tile_dim0 = 32
    g1[4] = 128;                  // tile_dim1 = 128, tile_dim2 = 0
    g1[5] = K;                    // tensor_dim0_stride[31:0] = K
    g1[6] = 0;                    // stride hi | tensor_dim1_stride lo
    g1[7] = 0;
    v4i z4 = {0, 0, 0, 0};
#if defined(__clang_major__) && (__clang_major__ >= 23)
    v8i z8 = {0, 0, 0, 0, 0, 0, 0, 0};
    __builtin_amdgcn_tensor_load_to_lds(g0, g1, z4, z4, z8, 0);
#else
    __builtin_amdgcn_tensor_load_to_lds(g0, g1, z4, z4, 0);
#endif
}

// ---------------------------------------------------------------------------
// fp32 -> bf16 bulk conversion (per-GEMM weight staging into ping buffer)
// ---------------------------------------------------------------------------
__global__ void k_cvt_bf16(bf16_t* __restrict__ dst, const float* __restrict__ src,
                           int n) {
    int i = blockIdx.x * 256 + threadIdx.x;
    if (i < n) dst[i] = (bf16_t)src[i];
}

// ---------------------------------------------------------------------------
// h = x + pos_embed[l]
// ---------------------------------------------------------------------------
__global__ void k_add_pos(float* __restrict__ h, const float* __restrict__ x,
                          const float* __restrict__ pos) {
    size_t i = (size_t)blockIdx.x * 256 + threadIdx.x;
    int l = (int)((i >> 9) % 100u);
    int d = (int)(i & 511u);
    h[i] = x[i] + pos[l * 512 + d];
}

// ---------------------------------------------------------------------------
// GEMM: C[M,N] = act(A[M,K] @ W[N,K]^T + bias[N])
// A fp32 (converted during LDS staging), W pre-converted bf16 (TDM-staged),
// fp32 accumulation. Block: 256 threads (8 waves), tile 128x128, K-step 32.
// Wave w = wy*2+wx computes m-tiles {wy*2, wy*2+1} x n-tiles {wx*4 .. wx*4+3}
// -> 8 WMMAs per K-step from 2 A-frags + 4 B-frags.
// M,N multiples of 128; K multiple of 32.
// ---------------------------------------------------------------------------
#define GLDA 48   // padded LDS K-stride (bf16 units): 96B rows, 16B aligned

template <int ACT>
__global__ __launch_bounds__(256)
void k_gemm(float* __restrict__ C, const float* __restrict__ A,
            const bf16_t* __restrict__ Wb, const float* __restrict__ bias,
            int M, int N, int K) {
    __shared__ __attribute__((aligned(16))) bf16_t As[128 * GLDA];
    __shared__ __attribute__((aligned(16))) bf16_t Ws[128 * GLDA];

    const int tid  = threadIdx.x;
    const int wave = tid >> 5;
    const int lane = tid & 31;
    const int half = lane >> 4;
    const int lr   = lane & 15;
    const int wy   = wave >> 1;          // 0..3  -> m-tile pair
    const int wx   = wave & 1;           // 0..1  -> n-tile quad
    const int m0   = blockIdx.y * 128;
    const int n0   = blockIdx.x * 128;

    const unsigned ws_off = (unsigned)(uintptr_t)(&Ws[0]);

    v8f acc[2][4] = {};

    for (int k0 = 0; k0 < K; k0 += 32) {
        __syncthreads();
        if (wave == 0)
            tdm_load_w_tile(Wb + (size_t)n0 * K + k0, ws_off, K);
        // stage A tile 128x32 with fused fp32->bf16 conversion
        for (int idx = tid; idx < 128 * 32; idx += 256) {
            int r = idx >> 5, c = idx & 31;
            As[r * GLDA + c] = (bf16_t)A[(size_t)(m0 + r) * K + k0 + c];
        }
        if (wave == 0)
            __builtin_amdgcn_s_wait_tensorcnt(0);
        __syncthreads();

        v16bf af[2], bfr[4];
#pragma unroll
        for (int i = 0; i < 2; i++)
            af[i] = load_frag_a(As + ((wy * 2 + i) * 16 + lr) * GLDA, half);
#pragma unroll
        for (int j = 0; j < 4; j++)
            bfr[j] = load_frag_b(Ws + ((wx * 4 + j) * 16 + lr) * GLDA, half);
#pragma unroll
        for (int i = 0; i < 2; i++)
#pragma unroll
            for (int j = 0; j < 4; j++)
                acc[i][j] = wmma_bf16(af[i], bfr[j], acc[i][j]);
    }

#pragma unroll
    for (int j = 0; j < 4; j++) {
        const int col = n0 + (wx * 4 + j) * 16 + lr;
        const float bv = bias ? bias[col] : 0.0f;
#pragma unroll
        for (int i = 0; i < 2; i++) {
#pragma unroll
            for (int r = 0; r < 8; r++) {
                int row = m0 + (wy * 2 + i) * 16 + half * 8 + r;
                float v = acc[i][j][r] + bv;
                if (ACT) v = gelu_exact(v);
                C[(size_t)row * N + col] = v;
            }
        }
    }
}

// ---------------------------------------------------------------------------
// Fused attention per (b,h):  S = QK^T + rel;  A = softmax(S/8);  O = A V
// Q,K,V,O layout: [B, L, H, E] fp32.  L=100 padded to 112; K-dim of A@V padded
// to 128.  Dynamic LDS: 132096 bytes (< 320KB WGP LDS).
// ---------------------------------------------------------------------------
#define ATTN_SMEM 132096
#define LDQ 72     // Qs/Ks K-stride (bf16): 144B rows
#define LDV 136    // Vt / Ab K-stride (bf16): 272B rows
#define LDSF 116   // scores stride (fp32)

__global__ __launch_bounds__(128)
void k_attn(float* __restrict__ O, const float* __restrict__ Q,
            const float* __restrict__ Km, const float* __restrict__ V,
            const float* __restrict__ rel /* [8,100] */) {
    extern __shared__ char smem[];
    bf16_t* Qs = (bf16_t*)(smem);                 // 112*72*2  = 16128
    bf16_t* Ks = (bf16_t*)(smem + 16128);         // 112*72*2  = 16128
    bf16_t* Vt = (bf16_t*)(smem + 32256);         // 64*136*2  = 17408
    float*  Sf = (float*)(smem + 49664);          // 112*116*4 = 51968
    bf16_t* Ab = (bf16_t*)(smem + 101632);        // 112*136*2 = 30464 -> 132096

    const int tid  = threadIdx.x;
    const int wave = tid >> 5;
    const int lane = tid & 31;
    const int half = lane >> 4;
    const int lr   = lane & 15;

    const int bh = blockIdx.x;         // b*8 + h
    const int b  = bh >> 3;
    const int h  = bh & 7;
    const size_t base = (size_t)b * 100 * 512 + h * 64;
    const float* relh = rel + h * 100;

    for (int idx = tid; idx < 112 * 64; idx += 128) {
        int l = idx >> 6, e = idx & 63;
        float qv = 0.0f, kv = 0.0f;
        if (l < 100) {
            qv = Q[base + (size_t)l * 512 + e];
            kv = Km[base + (size_t)l * 512 + e];
        }
        Qs[l * LDQ + e] = (bf16_t)qv;
        Ks[l * LDQ + e] = (bf16_t)kv;
    }
    for (int idx = tid; idx < 64 * 128; idx += 128) {
        int e = idx >> 7, s = idx & 127;
        float vv = (s < 100) ? V[base + (size_t)s * 512 + e] : 0.0f;
        Vt[e * LDV + s] = (bf16_t)vv;
    }
    __syncthreads();

    // GEMM1: S[l][s] = sum_e Q[l][e] K[s][e] (+ rel[s]); 7x7 tiles, K=64
    for (int t = wave; t < 49; t += 4) {
        int mt = t / 7, nt = t % 7;
        v8f acc = {};
#pragma unroll
        for (int ks = 0; ks < 64; ks += 32) {
            v16bf af = load_frag_a(Qs + (mt * 16 + lr) * LDQ + ks, half);
            v16bf bf = load_frag_b(Ks + (nt * 16 + lr) * LDQ + ks, half);
            acc = wmma_bf16(af, bf, acc);
        }
        int col = nt * 16 + lr;
        float relv = (col < 100) ? relh[col] : 0.0f;
#pragma unroll
        for (int j = 0; j < 8; j++) {
            int row = mt * 16 + half * 8 + j;
            Sf[row * LDSF + col] = acc[j] + relv;
        }
    }
    __syncthreads();

    // softmax over s (scale 1/sqrt(64) applied after rel-bias, as in reference)
    if (tid < 112) {
        int r = tid;
        if (r < 100) {
            float mx = -1e30f;
            for (int s = 0; s < 100; s++)
                mx = fmaxf(mx, Sf[r * LDSF + s] * 0.125f);
            float sum = 0.0f;
            for (int s = 0; s < 100; s++) {
                float e = __expf(Sf[r * LDSF + s] * 0.125f - mx);
                Sf[r * LDSF + s] = e;
                sum += e;
            }
            float inv = 1.0f / sum;
            for (int s = 0; s < 128; s++) {
                float v = (s < 100) ? Sf[r * LDSF + s] * inv : 0.0f;
                Ab[r * LDV + s] = (bf16_t)v;
            }
        } else {
            for (int s = 0; s < 128; s++) Ab[r * LDV + s] = (bf16_t)0.0f;
        }
    }
    __syncthreads();

    // GEMM2: O[l][e] = sum_s A[l][s] V[s][e]; 7x4 tiles, K=128
    for (int t = wave; t < 28; t += 4) {
        int mt = t >> 2, nt = t & 3;
        v8f acc = {};
#pragma unroll
        for (int ks = 0; ks < 128; ks += 32) {
            v16bf af = load_frag_a(Ab + (mt * 16 + lr) * LDV + ks, half);
            v16bf bf = load_frag_b(Vt + (nt * 16 + lr) * LDV + ks, half);
            acc = wmma_bf16(af, bf, acc);
        }
        int e = nt * 16 + lr;
#pragma unroll
        for (int j = 0; j < 8; j++) {
            int row = mt * 16 + half * 8 + j;
            if (row < 100)
                O[base + (size_t)row * 512 + e] = acc[j];
        }
    }
}

// ---------------------------------------------------------------------------
// Y[row] = LN(X[row] (+ R[row])) * G + Bb ; one wave per row, D=512.
// ---------------------------------------------------------------------------
__global__ __launch_bounds__(256)
void k_ln(float* __restrict__ Y, const float* __restrict__ X,
          const float* __restrict__ R, const float* __restrict__ G,
          const float* __restrict__ Bb) {
    int wave = threadIdx.x >> 5, lane = threadIdx.x & 31;
    size_t row = (size_t)blockIdx.x * 8 + wave;
    const float* x = X + row * 512;
    float vals[16];
    float sum = 0.0f;
#pragma unroll
    for (int i = 0; i < 16; i++) {
        float v = x[lane + i * 32];
        if (R) v += R[row * 512 + lane + i * 32];
        vals[i] = v;
        sum += v;
    }
    for (int off = 16; off; off >>= 1) sum += __shfl_xor(sum, off, 32);
    float mean = sum * (1.0f / 512.0f);
    float var = 0.0f;
#pragma unroll
    for (int i = 0; i < 16; i++) { float d = vals[i] - mean; var += d * d; }
    for (int off = 16; off; off >>= 1) var += __shfl_xor(var, off, 32);
    float rstd = rsqrtf(var * (1.0f / 512.0f) + 1e-5f);
    float* y = Y + row * 512;
#pragma unroll
    for (int i = 0; i < 16; i++) {
        int c = lane + i * 32;
        y[c] = (vals[i] - mean) * rstd * G[c] + Bb[c];
    }
}

// ---------------------------------------------------------------------------
// out[row] = dot(X[row], w) + b0 ; OUT=1, one wave per row.
// ---------------------------------------------------------------------------
__global__ __launch_bounds__(256)
void k_proj(float* __restrict__ out, const float* __restrict__ X,
            const float* __restrict__ w, const float* __restrict__ b0) {
    int wave = threadIdx.x >> 5, lane = threadIdx.x & 31;
    size_t row = (size_t)blockIdx.x * 8 + wave;
    const float* x = X + row * 512;
    float s = 0.0f;
#pragma unroll
    for (int i = 0; i < 16; i++) s += x[lane + i * 32] * w[lane + i * 32];
    for (int off = 16; off; off >>= 1) s += __shfl_xor(s, off, 32);
    if (lane == 0) out[row] = s + b0[0];
}

// ---------------------------------------------------------------------------
// host orchestration
// ---------------------------------------------------------------------------
extern "C" void kernel_launch(void* const* d_in, const int* in_sizes, int n_in,
                              void* d_out, int out_size, void* d_ws, size_t ws_size,
                              hipStream_t stream) {
    const int D = 512, DFF = 2048;
    const size_t BLD = (size_t)256 * 100 * 512;   // 13,107,200
    const int M = 25600;                           // B*L

    const float* x   = (const float*)d_in[0];
    const float* pos = (const float*)d_in[1];
    const float *eWq = (const float*)d_in[2],  *eWk = (const float*)d_in[3];
    const float *eWv = (const float*)d_in[4],  *eWo = (const float*)d_in[5];
    const float *ebq = (const float*)d_in[6],  *ebk = (const float*)d_in[7];
    const float *ebv = (const float*)d_in[8],  *ebo = (const float*)d_in[9];
    const float *erel = (const float*)d_in[10];
    const float *ec1w = (const float*)d_in[11], *ec1b = (const float*)d_in[12];
    const float *ec2w = (const float*)d_in[13], *ec2b = (const float*)d_in[14];
    const float *eln1g = (const float*)d_in[15], *eln1b = (const float*)d_in[16];
    const float *eln2g = (const float*)d_in[17], *eln2b = (const float*)d_in[18];
    const float *sWq = (const float*)d_in[19], *sWk = (const float*)d_in[20];
    const float *sWv = (const float*)d_in[21], *sWo = (const float*)d_in[22];
    const float *sbq = (const float*)d_in[23], *sbk = (const float*)d_in[24];
    const float *sbv = (const float*)d_in[25], *sbo = (const float*)d_in[26];
    const float *srel = (const float*)d_in[27];
    const float *cWq = (const float*)d_in[28], *cWk = (const float*)d_in[29];
    const float *cWv = (const float*)d_in[30], *cWo = (const float*)d_in[31];
    const float *cbq = (const float*)d_in[32], *cbk = (const float*)d_in[33];
    const float *cbv = (const float*)d_in[34], *cbo = (const float*)d_in[35];
    const float *crel = (const float*)d_in[36];
    const float *dc1w = (const float*)d_in[37], *dc1b = (const float*)d_in[38];
    const float *dc2w = (const float*)d_in[39], *dc2b = (const float*)d_in[40];
    const float *dln1g = (const float*)d_in[41], *dln1b = (const float*)d_in[42];
    const float *dln2g = (const float*)d_in[43], *dln2b = (const float*)d_in[44];
    const float *dln3g = (const float*)d_in[45], *dln3b = (const float*)d_in[46];
    const float *encg = (const float*)d_in[47], *encb = (const float*)d_in[48];
    const float *decg = (const float*)d_in[49], *decb = (const float*)d_in[50];
    const float *pw = (const float*)d_in[51], *pb = (const float*)d_in[52];

    float* h     = (float*)d_ws;
    float* cross = h + BLD;
    float* t1    = h + 2 * BLD;
    float* q     = h + 3 * BLD;
    float* kk    = h + 4 * BLD;
    float* vv    = h + 5 * BLD;
    float* ao    = h + 6 * BLD;
    float* ffh   = q;                       // reuse q..ao: 4*BLD == B*L*DFF
    bf16_t* wbuf = (bf16_t*)(h + 7 * BLD);  // bf16 weight ping buffer (max D*DFF)

    auto gemm = [&](float* C, const float* A, const float* W, const float* bias,
                    int N, int K, bool act) {
        int n = N * K;
        k_cvt_bf16<<<(n + 255) / 256, 256, 0, stream>>>(wbuf, W, n);
        dim3 grid(N / 128, M / 128);
        if (act) k_gemm<1><<<grid, 256, 0, stream>>>(C, A, wbuf, bias, M, N, K);
        else     k_gemm<0><<<grid, 256, 0, stream>>>(C, A, wbuf, bias, M, N, K);
    };

    k_add_pos<<<51200, 256, 0, stream>>>(h, x, pos);

    // ---------------- encoder ----------------
    for (int i = 0; i < 4; i++) {
        const size_t wo = (size_t)i * D * D, bo_ = (size_t)i * D;
        gemm(q,  h, eWq + wo, ebq + bo_, D, D, false);
        gemm(kk, h, eWk + wo, ebk + bo_, D, D, false);
        gemm(vv, h, eWv + wo, ebv + bo_, D, D, false);
        k_attn<<<2048, 128, ATTN_SMEM, stream>>>(ao, q, kk, vv, erel + (size_t)i * 800);
        gemm(t1, ao, eWo + wo, ebo + bo_, D, D, false);
        k_ln<<<3200, 256, 0, stream>>>(h, h, t1, eln1g + bo_, eln1b + bo_);
        gemm(ffh, h, ec1w + (size_t)i * DFF * D, ec1b + (size_t)i * DFF, DFF, D, true);
        gemm(t1, ffh, ec2w + (size_t)i * D * DFF, ec2b + bo_, D, DFF, false);
        k_ln<<<3200, 256, 0, stream>>>(h, h, t1, eln2g + bo_, eln2b + bo_);
    }
    k_ln<<<3200, 256, 0, stream>>>(h, h, nullptr, encg, encb);
    hipMemcpyAsync(cross, h, BLD * sizeof(float), hipMemcpyDeviceToDevice, stream);

    // ---------------- decoder ----------------
    for (int i = 0; i < 3; i++) {
        const size_t wo = (size_t)i * D * D, bo_ = (size_t)i * D;
        gemm(q,  h, sWq + wo, sbq + bo_, D, D, false);
        gemm(kk, h, sWk + wo, sbk + bo_, D, D, false);
        gemm(vv, h, sWv + wo, sbv + bo_, D, D, false);
        k_attn<<<2048, 128, ATTN_SMEM, stream>>>(ao, q, kk, vv, srel + (size_t)i * 800);
        gemm(t1, ao, sWo + wo, sbo + bo_, D, D, false);
        k_ln<<<3200, 256, 0, stream>>>(h, h, t1, dln1g + bo_, dln1b + bo_);
        gemm(q,  h,     cWq + wo, cbq + bo_, D, D, false);
        gemm(kk, cross, cWk + wo, cbk + bo_, D, D, false);
        gemm(vv, cross, cWv + wo, cbv + bo_, D, D, false);
        k_attn<<<2048, 128, ATTN_SMEM, stream>>>(ao, q, kk, vv, crel + (size_t)i * 800);
        gemm(t1, ao, cWo + wo, cbo + bo_, D, D, false);
        k_ln<<<3200, 256, 0, stream>>>(h, h, t1, dln2g + bo_, dln2b + bo_);
        gemm(ffh, h, dc1w + (size_t)i * DFF * D, dc1b + (size_t)i * DFF, DFF, D, true);
        gemm(t1, ffh, dc2w + (size_t)i * D * DFF, dc2b + bo_, D, DFF, false);
        k_ln<<<3200, 256, 0, stream>>>(h, h, t1, dln3g + bo_, dln3b + bo_);
    }
    k_ln<<<3200, 256, 0, stream>>>(h, h, nullptr, decg, decb);

    k_proj<<<3200, 256, 0, stream>>>((float*)d_out, h, pw, pb);

    (void)in_sizes; (void)n_in; (void)out_size; (void)ws_size;
}
